// FlashAttention_71751723647185
// MI455X (gfx1250) — matrix operand order repaired and checked by reading, not compiled, not run on hardware
//
#include <hip/hip_runtime.h>
#include <hip/hip_bf16.h>

// ---------------------------------------------------------------------------
// Problem constants (from the reference)
// ---------------------------------------------------------------------------
#define BATCH   4
#define SEQ     2048
#define DMODEL  1024
#define HEADS   16
#define DK      64            // DMODEL / HEADS
#define NTOK    (BATCH * SEQ) // 8192
#define D3      (3 * DMODEL)  // 3072

typedef __attribute__((ext_vector_type(16))) __bf16 v16bf;
typedef __attribute__((ext_vector_type(8)))  __bf16 v8bf;
typedef __attribute__((ext_vector_type(8)))  float  v8f;

typedef __attribute__((ext_vector_type(4))) unsigned int u32x4;
typedef __attribute__((ext_vector_type(8))) int          i32x8;
typedef __attribute__((ext_vector_type(4))) int          i32x4;

#if __has_builtin(__builtin_amdgcn_tensor_load_to_lds)
#define HAVE_TDM 1
#else
#define HAVE_TDM 0
#endif

// ---------------------------------------------------------------------------
// Helpers
// ---------------------------------------------------------------------------
__device__ __forceinline__ unsigned short f2bf(float f) {
    unsigned u = __float_as_uint(f);
    unsigned r = (u + 0x7FFFu + ((u >> 16) & 1u)) >> 16;   // round-nearest-even
    return (unsigned short)r;
}

// Build a 16-element bf16 fragment from two 16-byte chunks (lowers to b128 loads)
__device__ __forceinline__ v16bf ld_frag(const unsigned short* p0,
                                         const unsigned short* p1) {
    v8bf lo = *reinterpret_cast<const v8bf*>(p0);
    v8bf hi = *reinterpret_cast<const v8bf*>(p1);
    v16bf r;
#pragma unroll
    for (int i = 0; i < 8; ++i) { r[i] = lo[i]; r[8 + i] = hi[i]; }
    return r;
}

__device__ __forceinline__ v8f wmma_bf16(v16bf a, v16bf b, v8f c) {
    return __builtin_amdgcn_wmma_f32_16x16x32_bf16(
        false, a, false, b, (short)0, c, false, false);
}

// ---- DPP16 butterfly reductions across a 16-lane row (no LDS traffic) ------
// xor1 = quad_perm(1,0,3,2)=0xB1; xor2 = quad_perm(2,3,0,1)=0x4E;
// "xor4" = row_half_mirror (0x141) after quads are uniform;
// "xor8" = row_mirror (0x140) after 8-groups are uniform.
template <int CTRL>
__device__ __forceinline__ float dpp_xfer(float x) {
    return __int_as_float(__builtin_amdgcn_update_dpp(
        0, __float_as_int(x), CTRL, 0xF, 0xF, true));
}
__device__ __forceinline__ float rowmax16(float x) {
    x = fmaxf(x, dpp_xfer<0xB1>(x));
    x = fmaxf(x, dpp_xfer<0x4E>(x));
    x = fmaxf(x, dpp_xfer<0x141>(x));
    x = fmaxf(x, dpp_xfer<0x140>(x));
    return x;
}
__device__ __forceinline__ float rowsum16(float x) {
    x += dpp_xfer<0xB1>(x);
    x += dpp_xfer<0x4E>(x);
    x += dpp_xfer<0x141>(x);
    x += dpp_xfer<0x140>(x);
    return x;
}

// ---------------------------------------------------------------------------
// TDM: issue a 2D tile load (tile_d0 x tile_d1 elements of 2 bytes) from a
// row-major tensor (row stride = stride_elems) into LDS at lds_off, with
// optional LDS row padding (pad encodings per D# spec).
// ---------------------------------------------------------------------------
#if HAVE_TDM
__device__ __forceinline__ void tdm_load_2d_bf16(
    unsigned lds_off, unsigned long long gaddr_bytes,
    unsigned tile_d0, unsigned tile_d1,
    unsigned long long stride_elems,
    unsigned tensor_d0, unsigned tensor_d1,
    int pad_interval, int pad_amount, bool pad_en) {
    u32x4 g0;
    g0[0] = 1u;                                   // count=1 (valid user D#)
    g0[1] = lds_off;                              // lds_addr (bytes)
    g0[2] = (unsigned)gaddr_bytes;                // global_addr[31:0]
    g0[3] = (unsigned)((gaddr_bytes >> 32) & 0x01FFFFFFull) | (2u << 30); // type=2
    unsigned w0 = (1u << 16);                     // data_size=1 -> 2 bytes
    if (pad_en)
        w0 |= (1u << 20) | ((unsigned)pad_interval << 22) |
              ((unsigned)pad_amount << 25);
    i32x8 g1;
    g1[0] = (int)w0;                                                     // mask=0
    g1[1] = (int)((tensor_d0 & 0xFFFFu) << 16);                          // dim0 lo
    g1[2] = (int)(((tensor_d0 >> 16) & 0xFFFFu) | ((tensor_d1 & 0xFFFFu) << 16));
    g1[3] = (int)(((tensor_d1 >> 16) & 0xFFFFu) | ((tile_d0 & 0xFFFFu) << 16));
    g1[4] = (int)(tile_d1 & 0xFFFFu);                                    // tile_dim1
    g1[5] = (int)(unsigned)(stride_elems & 0xFFFFFFFFull);               // stride0 lo
    g1[6] = (int)(unsigned)((stride_elems >> 32) & 0xFFFFull);           // stride0 hi
    g1[7] = 0;
    i32x4 z4 = {0, 0, 0, 0};
#if defined(__clang_major__) && __clang_major__ >= 23
    i32x8 z8 = {0, 0, 0, 0, 0, 0, 0, 0};
    __builtin_amdgcn_tensor_load_to_lds(g0, g1, z4, z4, z8, 0);
#else
    __builtin_amdgcn_tensor_load_to_lds(g0, g1, z4, z4, 0);
#endif
}
#endif

__device__ __forceinline__ unsigned lds_offset_of(const void* p) {
    return (unsigned)(unsigned long long)(uintptr_t)p;  // low 32 bits = LDS offset
}

// ---------------------------------------------------------------------------
// Kernel 0: f32 -> bf16 conversion (grid-stride)
// ---------------------------------------------------------------------------
__global__ void convert_f32_to_bf16(const float* __restrict__ in,
                                    unsigned short* __restrict__ out, int n) {
    int i = blockIdx.x * blockDim.x + threadIdx.x;
    int stride = gridDim.x * blockDim.x;
    for (; i < n; i += stride) out[i] = f2bf(in[i]);
}

// ---------------------------------------------------------------------------
// Kernel 1/3: bf16 WMMA GEMM  C[M,N] = A[M,K] * B[N,K]^T
//   256 threads (8 waves); block tile 128(M) x 64(N); K slab 32.
//   TDM path: wave 0 DMA-stages both tiles (double-buffered, TENSORcnt),
//   pad fields give the 40-half padded LDS row stride for conflict-free b128s.
// ---------------------------------------------------------------------------
template <bool OUT_F32>
__global__ __launch_bounds__(256)
void gemm_bf16_wmma(const unsigned short* __restrict__ A,
                    const unsigned short* __restrict__ Bw,
                    void* __restrict__ out,
                    int M, int N, int K) {
    constexpr int ST = 40;  // 32 + 8 pad halves; 80 B rows (16B aligned)
    __shared__ __align__(16) unsigned short sA[2][128 * ST];
    __shared__ __align__(16) unsigned short sB[2][64 * ST];

    const int tid   = threadIdx.x;
    const int wave  = tid >> 5;
    const int lane  = tid & 31;
    const int lm    = lane & 15;
    const int hi    = lane >> 4;
    const int waveM = wave & 3;
    const int waveN = wave >> 2;

    const int rowBase = blockIdx.y * 128;
    const int colBase = blockIdx.x * 64;
    const int nsteps  = K >> 5;

    v8f acc[2][2];
#pragma unroll
    for (int i = 0; i < 2; ++i)
#pragma unroll
        for (int j = 0; j < 2; ++j)
#pragma unroll
            for (int r = 0; r < 8; ++r) acc[i][j][r] = 0.0f;

    const int khalfA = hi ? 8 : 0;
    const int kbB    = hi ? 8 : 0;

#if HAVE_TDM
    // pad: 16 DW (one 32-half row) -> interval code 3; +4 DW -> amount code 3
    if (wave == 0) {
        tdm_load_2d_bf16(lds_offset_of(&sA[0][0]),
                         (unsigned long long)(uintptr_t)A +
                             ((unsigned long long)rowBase * K) * 2ull,
                         32, 128, (unsigned long long)K, 32, 128, 3, 3, true);
        tdm_load_2d_bf16(lds_offset_of(&sB[0][0]),
                         (unsigned long long)(uintptr_t)Bw +
                             ((unsigned long long)colBase * K) * 2ull,
                         32, 64, (unsigned long long)K, 32, 64, 3, 3, true);
    }
    for (int ks = 0; ks < nsteps; ++ks) {
        const int cur = ks & 1;
        if (wave == 0) {
            if (ks + 1 < nsteps) {
                const int k0n = (ks + 1) << 5;
                tdm_load_2d_bf16(lds_offset_of(&sA[cur ^ 1][0]),
                                 (unsigned long long)(uintptr_t)A +
                                     ((unsigned long long)rowBase * K + k0n) * 2ull,
                                 32, 128, (unsigned long long)K, 32, 128, 3, 3, true);
                tdm_load_2d_bf16(lds_offset_of(&sB[cur ^ 1][0]),
                                 (unsigned long long)(uintptr_t)Bw +
                                     ((unsigned long long)colBase * K + k0n) * 2ull,
                                 32, 64, (unsigned long long)K, 32, 64, 3, 3, true);
                __builtin_amdgcn_s_wait_tensorcnt(2);  // current pair landed
            } else {
                __builtin_amdgcn_s_wait_tensorcnt(0);
            }
        }
        __syncthreads();
        v16bf af[2], bf[2];
#pragma unroll
        for (int mf = 0; mf < 2; ++mf) {
            const unsigned short* p =
                &sA[cur][(waveM * 32 + mf * 16 + lm) * ST + khalfA];
            af[mf] = ld_frag(p, p + 16);
        }
#pragma unroll
        for (int nf = 0; nf < 2; ++nf) {
            const unsigned short* p =
                &sB[cur][(waveN * 32 + nf * 16 + lm) * ST + kbB];
            bf[nf] = ld_frag(p, p + 16);
        }
#pragma unroll
        for (int mf = 0; mf < 2; ++mf)
#pragma unroll
            for (int nf = 0; nf < 2; ++nf)
                acc[mf][nf] = wmma_bf16(af[mf], bf[nf], acc[mf][nf]);
        __syncthreads();  // all reads of buf[cur] done before it is re-filled
    }
#else
    for (int ks = 0; ks < nsteps; ++ks) {
        const int k0 = ks << 5;
        __syncthreads();
        {
            int r  = tid >> 1;
            int cs = tid & 1;
            const uint4* g = reinterpret_cast<const uint4*>(
                A + (size_t)(rowBase + r) * K + k0 + cs * 16);
            uint4* d = reinterpret_cast<uint4*>(&sA[0][r * ST + cs * 16]);
            d[0] = g[0];
            d[1] = g[1];
        }
        {
            int r  = tid >> 2;
            int cs = tid & 3;
            const uint4* g = reinterpret_cast<const uint4*>(
                Bw + (size_t)(colBase + r) * K + k0 + cs * 8);
            uint4* d = reinterpret_cast<uint4*>(&sB[0][r * ST + cs * 8]);
            d[0] = g[0];
        }
        __syncthreads();
        v16bf af[2], bf[2];
#pragma unroll
        for (int mf = 0; mf < 2; ++mf) {
            const unsigned short* p =
                &sA[0][(waveM * 32 + mf * 16 + lm) * ST + khalfA];
            af[mf] = ld_frag(p, p + 16);
        }
#pragma unroll
        for (int nf = 0; nf < 2; ++nf) {
            const unsigned short* p =
                &sB[0][(waveN * 32 + nf * 16 + lm) * ST + kbB];
            bf[nf] = ld_frag(p, p + 16);
        }
#pragma unroll
        for (int mf = 0; mf < 2; ++mf)
#pragma unroll
            for (int nf = 0; nf < 2; ++nf)
                acc[mf][nf] = wmma_bf16(af[mf], bf[nf], acc[mf][nf]);
    }
#endif

#pragma unroll
    for (int mf = 0; mf < 2; ++mf)
#pragma unroll
        for (int nf = 0; nf < 2; ++nf)
#pragma unroll
            for (int r = 0; r < 8; ++r) {
                int grow = rowBase + waveM * 32 + mf * 16 + r + 8 * hi;
                int gcol = colBase + waveN * 32 + nf * 16 + lm;
                float v = acc[mf][nf][r];
                if (OUT_F32)
                    reinterpret_cast<float*>(out)[(size_t)grow * N + gcol] = v;
                else
                    reinterpret_cast<unsigned short*>(out)[(size_t)grow * N + gcol] =
                        f2bf(v);
            }
}

// ---------------------------------------------------------------------------
// Kernel 2: flash attention (non-causal) over bf16 QKV
//   grid = (T/64, B*H); block = 128 threads (4 waves x 16 q-rows).
//   K tile staged by TDM (padded rows); V transposed on store by all threads;
//   softmax row reductions entirely in DPP VALU.
// ---------------------------------------------------------------------------
__global__ __launch_bounds__(128)
void flash_attn(const unsigned short* __restrict__ qkv,
                unsigned short* __restrict__ ao) {
    constexpr int SK_STRIDE = 72;  // 64 + 8 pad halves, 144 B rows
    constexpr int SV_STRIDE = 40;  // transposed [dk][key]
    constexpr int SP_STRIDE = 40;
    __shared__ __align__(16) unsigned short sK[32 * SK_STRIDE];
    __shared__ __align__(16) unsigned short sV[64 * SV_STRIDE];
    __shared__ __align__(16) unsigned short sP[4][16 * SP_STRIDE];

    const int bh   = blockIdx.y;
    const int b    = bh >> 4;
    const int h    = bh & 15;
    const int wave = threadIdx.x >> 5;
    const int lane = threadIdx.x & 31;
    const int lm   = lane & 15;
    const int hi   = lane >> 4;
    const int khalf = hi ? 8 : 0;

    const int t0 = blockIdx.x * 64 + wave * 16;
    const float scale = 0.125f;   // 1/sqrt(64)

    // ---- Q fragments, pre-scaled by 1/sqrt(dk), resident in VGPRs ----
    v16bf qf[2];
    {
        size_t qbase = ((size_t)(b * SEQ + t0 + lm)) * D3 + h * DK;
#pragma unroll
        for (int ks = 0; ks < 2; ++ks) {
            const unsigned short* p = qkv + qbase + ks * 32 + khalf;
            v16bf q = ld_frag(p, p + 16);
#pragma unroll
            for (int i = 0; i < 16; ++i)
                q[i] = (__bf16)((float)q[i] * scale);
            qf[ks] = q;
        }
    }

    v8f o[4];
#pragma unroll
    for (int g = 0; g < 4; ++g)
#pragma unroll
        for (int r = 0; r < 8; ++r) o[g][r] = 0.0f;
    float m[8], l[8];
#pragma unroll
    for (int r = 0; r < 8; ++r) { m[r] = -1e30f; l[r] = 0.0f; }

    for (int kt = 0; kt < SEQ / 32; ++kt) {
        const int key0 = kt * 32;
        __syncthreads();   // prior tile fully consumed
#if HAVE_TDM
        if (wave == 0) {
            // K tile: 64 (dk) x 32 (keys); row stride D3; pad 32DW rows by 4DW
            unsigned long long gk =
                (unsigned long long)(uintptr_t)qkv +
                (((unsigned long long)(b * SEQ + key0)) * D3 + DMODEL + h * DK) * 2ull;
            tdm_load_2d_bf16(lds_offset_of(sK), gk, 64, 32,
                             (unsigned long long)D3, 64, 32, 4, 3, true);
        }
#else
        {
            int r  = threadIdx.x >> 2;
            int cs = threadIdx.x & 3;
            size_t krow = ((size_t)(b * SEQ + key0 + r)) * D3 + DMODEL + h * DK;
            const uint4* gk = reinterpret_cast<const uint4*>(qkv + krow + cs * 16);
            uint4* dk = reinterpret_cast<uint4*>(sK + r * SK_STRIDE + cs * 16);
            dk[0] = gk[0];
            dk[1] = gk[1];
        }
#endif
        // ---- V tile transposed on store: [dk][key] ----
        {
            int r  = threadIdx.x >> 2;   // key 0..31
            int cs = threadIdx.x & 3;    // 16-half dk segment
            size_t vrow = ((size_t)(b * SEQ + key0 + r)) * D3 + 2 * DMODEL + h * DK;
            const unsigned short* gv = qkv + vrow + cs * 16;
#pragma unroll
            for (int d = 0; d < 16; ++d)
                sV[(cs * 16 + d) * SV_STRIDE + r] = gv[d];
        }
#if HAVE_TDM
        if (wave == 0) __builtin_amdgcn_s_wait_tensorcnt(0);
#endif
        __syncthreads();

        // ---- S = Q * K^T (two 16-key column groups) ----
        v8f s0, s1;
#pragma unroll
        for (int r = 0; r < 8; ++r) { s0[r] = 0.0f; s1[r] = 0.0f; }
#pragma unroll
        for (int ks = 0; ks < 2; ++ks) {
            const unsigned short* kp0 = sK + lm * SK_STRIDE + ks * 32 + khalf;
            const unsigned short* kp1 = sK + (16 + lm) * SK_STRIDE + ks * 32 + khalf;
            s0 = wmma_bf16(qf[ks], ld_frag(kp0, kp0 + 16), s0);
            s1 = wmma_bf16(qf[ks], ld_frag(kp1, kp1 + 16), s1);
        }

        // ---- online softmax; DPP row reductions (no LDS) ----
#pragma unroll
        for (int r = 0; r < 8; ++r) {
            float a0 = s0[r];
            float a1 = s1[r];
            float rm = rowmax16(fmaxf(a0, a1));
            float mn = fmaxf(m[r], rm);
            float alpha = __expf(m[r] - mn);
            l[r] *= alpha;
#pragma unroll
            for (int g = 0; g < 4; ++g) o[g][r] *= alpha;
            float p0 = __expf(a0 - mn);
            float p1 = __expf(a1 - mn);
            l[r] += p0 + p1;
            m[r] = mn;
            int row = r + 8 * hi;   // C-layout row of this element
            sP[wave][row * SP_STRIDE + lm]      = f2bf(p0);
            sP[wave][row * SP_STRIDE + 16 + lm] = f2bf(p1);
        }
        asm volatile("s_wait_dscnt 0" ::: "memory");  // wave-local LDS RAW fence

        // ---- P (A-fragment 16x32) and O += P * V ----
        const unsigned short* pp = &sP[wave][lm * SP_STRIDE + khalf];
        v16bf pf = ld_frag(pp, pp + 16);
#pragma unroll
        for (int g = 0; g < 4; ++g) {
            const unsigned short* vp = sV + (g * 16 + lm) * SV_STRIDE + khalf;
            o[g] = wmma_bf16(pf, ld_frag(vp, vp + 16), o[g]);
        }
    }

    // ---- finalize ----
    float inv[8];
#pragma unroll
    for (int r = 0; r < 8; ++r) inv[r] = 1.0f / rowsum16(l[r]);
#pragma unroll
    for (int g = 0; g < 4; ++g)
#pragma unroll
        for (int r = 0; r < 8; ++r) {
            int row = t0 + r + 8 * hi;
            int col = h * DK + g * 16 + lm;
            ao[((size_t)(b * SEQ + row)) * DMODEL + col] = f2bf(o[g][r] * inv[r]);
        }
}

// ---------------------------------------------------------------------------
// Launcher
// ---------------------------------------------------------------------------
extern "C" void kernel_launch(void* const* d_in, const int* in_sizes, int n_in,
                              void* d_out, int out_size, void* d_ws, size_t ws_size,
                              hipStream_t stream) {
    const float* X     = (const float*)d_in[0];
    const float* W_qkv = (const float*)d_in[1];
    const float* W_o   = (const float*)d_in[2];
    float* out = (float*)d_out;

    unsigned short* ws    = (unsigned short*)d_ws;
    unsigned short* Xb    = ws;
    unsigned short* Wqkvb = Xb    + (size_t)NTOK * DMODEL;
    unsigned short* Wob   = Wqkvb + (size_t)D3 * DMODEL;
    unsigned short* QKVb  = Wob   + (size_t)DMODEL * DMODEL;
    unsigned short* AOb   = QKVb  + (size_t)NTOK * D3;

    convert_f32_to_bf16<<<2048, 256, 0, stream>>>(X,     Xb,    NTOK * DMODEL);
    convert_f32_to_bf16<<<1024, 256, 0, stream>>>(W_qkv, Wqkvb, D3 * DMODEL);
    convert_f32_to_bf16<<<512,  256, 0, stream>>>(W_o,   Wob,   DMODEL * DMODEL);

    gemm_bf16_wmma<false><<<dim3(D3 / 64, NTOK / 128), 256, 0, stream>>>(
        Xb, Wqkvb, QKVb, NTOK, D3, DMODEL);

    flash_attn<<<dim3(SEQ / 64, BATCH * HEADS), 128, 0, stream>>>(QKVb, AOb);

    gemm_bf16_wmma<true><<<dim3(DMODEL / 64, NTOK / 128), 256, 0, stream>>>(
        AOb, Wob, out, NTOK, DMODEL, DMODEL);
}
